// CPC_19490561589294
// MI455X (gfx1250) — matrix operand order repaired and checked by reading, not compiled, not run on hardware
//
#include <hip/hip_runtime.h>
#include <hip/hip_bf16.h>

// ---------------------------------------------------------------------------
// InfoNCE loss, MI455X (gfx1250, wave32) version.
//   x_pred = y @ W^T + b ; rows L2-normalized; xn = normalize(x)
//   pos[n] = xn[n].x_pred_n[n];  neg[n] = logsumexp_m(xn[n].x_pred_n[m])
//   out    = mean(neg - pos)
// GEMM2 (137 GFLOP) runs on V_WMMA_F32_16X16X32_BF16 with fused exp-sum.
// Key property: all scores are cosine similarities (|s| <= 1), so the
// logsumexp max-shift is provably unnecessary -> hot loop is pure
// wmma + b128 loads + one v_exp per accumulator row (no cross-lane ops).
// ---------------------------------------------------------------------------

typedef __attribute__((ext_vector_type(16))) __bf16 v16bf;
typedef __attribute__((ext_vector_type(8)))  __bf16 v8bf;
typedef __attribute__((ext_vector_type(8)))  float  v8f;

#define NROWS  8192
#define DIM    1024
#define MSPLIT 8
#define MCHUNK (NROWS / MSPLIT)   // 1024 m-rows per grid.y slice

// ---- WMMA fragment loaders (bf16, 16x16x32, per CDNA5 ISA) ------
// A (16x32, MxK): lane L holds row M=L&15; lanes 0-15 carry K[0..7]+K[16..23],
// lanes 16-31 carry K[8..15]+K[24..31] (2 elems/VGPR).
static __device__ __forceinline__ v16bf load_a_frag(const __bf16* A, int lda,
                                                    int m0, int k0, int lane) {
    int row  = lane & 15;
    int half = lane >> 4;
    const __bf16* p = A + (size_t)(m0 + row) * lda + k0 + half * 8;
    union { v16bf v; v8bf h[2]; } u;
    u.h[0] = *(const v8bf*)(p);        // K chunk [half*8 .. half*8+7]
    u.h[1] = *(const v8bf*)(p + 16);   // K chunk [16+half*8 .. 16+half*8+7]
    return u.v;
}
// B (32x16, KxN), fed from row-major rows of the "other" matrix (A@B^T form):
// lane L holds column N=L&15; lanes 0-15 carry K[0..7]+K[16..23], lanes 16-31 K[8..15]+K[24..31].
static __device__ __forceinline__ v16bf load_b_frag(const __bf16* B, int ldb,
                                                    int n0, int k0, int lane) {
    int col  = lane & 15;
    int half = lane >> 4;
    const __bf16* p = B + (size_t)(n0 + col) * ldb + k0 + half * 8;
    union { v16bf v; v8bf h[2]; } u;
    u.h[0] = *(const v8bf*)(p);
    u.h[1] = *(const v8bf*)(p + 16);
    return u.v;
}

// ---- elementwise f32 -> bf16 ----------------------------------------------
__global__ void k_cvt_bf16(const float* __restrict__ in, __bf16* __restrict__ out, int count) {
    int stride = gridDim.x * blockDim.x;
    for (int i = blockIdx.x * blockDim.x + threadIdx.x; i < count; i += stride)
        out[i] = (__bf16)in[i];
}

// ---- row L2-normalize (optionally + bias), f32 in -> bf16 out -------------
__global__ void k_rownorm_bf16(const float* __restrict__ in, const float* __restrict__ bias,
                               int has_bias, __bf16* __restrict__ out) {
    __shared__ float red[256];
    int t = threadIdx.x;
    size_t rowbase = (size_t)blockIdx.x * DIM;
    float v[4];
    float ss = 0.f;
#pragma unroll
    for (int j = 0; j < 4; ++j) {
        int c = t * 4 + j;
        float xv = in[rowbase + c];
        if (has_bias) xv += bias[c];
        v[j] = xv;
        ss += xv * xv;
    }
    red[t] = ss;
    __syncthreads();
    for (int off = 128; off > 0; off >>= 1) {
        if (t < off) red[t] += red[t + off];
        __syncthreads();
    }
    float rn = rsqrtf(red[0]);
#pragma unroll
    for (int j = 0; j < 4; ++j)
        out[rowbase + t * 4 + j] = (__bf16)(v[j] * rn);
}

// ---- GEMM1: x_pred[m,n] = sum_k ybf[m,k]*Wbf[n,k] (one 16x16 tile / wave) --
__global__ void __launch_bounds__(256) k_gemm1_wmma(const __bf16* __restrict__ Y,
                                                    const __bf16* __restrict__ Wb,
                                                    float* __restrict__ XP) {
    int lane = threadIdx.x & 31;
    int wave = threadIdx.x >> 5;
    int tile = blockIdx.x * 8 + wave;        // 32768 tiles total
    int nt = tile & 63;                      // 64 n-tiles (DIM/16)
    int mt = tile >> 6;                      // 512 m-tiles
    int m0 = mt * 16, n0 = nt * 16;
    v8f acc = {};
#pragma unroll 4
    for (int k0 = 0; k0 < DIM; k0 += 32) {
        v16bf a = load_a_frag(Y, DIM, m0, k0, lane);
        v16bf b = load_b_frag(Wb, DIM, n0, k0, lane);
        acc = __builtin_amdgcn_wmma_f32_16x16x32_bf16(false, a, false, b,
                                                      (short)0, acc, false, false);
    }
    // C layout: VGPR r, lanes 0-15 -> M=r, lanes 16-31 -> M=r+8; N = lane&15
    int col = lane & 15, half = lane >> 4;
#pragma unroll
    for (int r = 0; r < 8; ++r)
        XP[(size_t)(m0 + r + 8 * half) * DIM + n0 + col] = acc[r];
}

// ---- pos[n] = dot(xn[n], xpredn[n]) (one wave per row) --------------------
__global__ void k_pos(const __bf16* __restrict__ xn, const __bf16* __restrict__ xp,
                      float* __restrict__ pos) {
    int lane = threadIdx.x & 31;
    int wave = threadIdx.x >> 5;
    int row = blockIdx.x * 8 + wave;
    const __bf16* a = xn + (size_t)row * DIM;
    const __bf16* b = xp + (size_t)row * DIM;
    float s = 0.f;
#pragma unroll 4
    for (int c = lane; c < DIM; c += 32)
        s += (float)a[c] * (float)b[c];
#pragma unroll
    for (int off = 16; off; off >>= 1) s += __shfl_xor(s, off, 32);
    if (lane == 0) pos[row] = s;
}

// ---- GEMM2 + fused exp-sum (scores are cosines: no max-shift needed) ------
// Wave owns 16 n-rows; A fragments for full K hoisted into 256 VGPRs.
// grid.y slices the m-range 8 ways; partial sums written per slice.
__global__ void __launch_bounds__(256) k_gemm2_lse(const __bf16* __restrict__ XN,
                                                   const __bf16* __restrict__ XPN,
                                                   float* __restrict__ psum) {
    int lane = threadIdx.x & 31;
    int wave = threadIdx.x >> 5;
    int n0   = blockIdx.x * 128 + wave * 16;
    int mbeg = blockIdx.y * MCHUNK;

    v16bf afrag[32];
#pragma unroll
    for (int kk = 0; kk < 32; ++kk)
        afrag[kk] = load_a_frag(XN, DIM, n0, kk * 32, lane);

    // per-(row, lane) partial sum of exp(score); lane covers cols m%16==col
    float rs[8];
#pragma unroll
    for (int r = 0; r < 8; ++r) rs[r] = 0.f;

    for (int mt = 0; mt < MCHUNK; mt += 16) {
        // prefetch next B tile's first cacheline (global_prefetch_b8)
        if (mt + 16 < MCHUNK)
            __builtin_prefetch(XPN + (size_t)(mbeg + mt + 16 + (lane & 15)) * DIM, 0, 3);
        v8f acc = {};
#pragma unroll
        for (int kk = 0; kk < 32; ++kk) {
            v16bf b = load_b_frag(XPN, DIM, mbeg + mt, kk * 32, lane);
            acc = __builtin_amdgcn_wmma_f32_16x16x32_bf16(false, afrag[kk], false, b,
                                                          (short)0, acc, false, false);
        }
#pragma unroll
        for (int r = 0; r < 8; ++r)
            rs[r] += __expf(acc[r]);     // |score|<=1 -> always safe
    }

    // one-time 16-lane tree sum within each half; row M = r + 8*(lane>=16)
    int col = lane & 15, half = lane >> 4;
#pragma unroll
    for (int r = 0; r < 8; ++r) {
        float s = rs[r];
        s += __shfl_xor(s, 8, 16);
        s += __shfl_xor(s, 4, 16);
        s += __shfl_xor(s, 2, 16);
        s += __shfl_xor(s, 1, 16);
        if (col == 0)
            psum[(size_t)blockIdx.y * NROWS + n0 + r + 8 * half] = s;
    }
}

// ---- merge m-slices: npm[n] = log(sum exp) - pos[n] -----------------------
__global__ void k_merge(const float* __restrict__ psum, const float* __restrict__ pos,
                        float* __restrict__ npm) {
    int n = blockIdx.x * blockDim.x + threadIdx.x;
    float S = 0.f;
#pragma unroll
    for (int s = 0; s < MSPLIT; ++s)
        S += psum[(size_t)s * NROWS + n];
    npm[n] = __logf(S) - pos[n];
}

// ---- two-stage mean reduction ---------------------------------------------
__global__ void k_reduce1(const float* __restrict__ npm, float* __restrict__ partial) {
    __shared__ float red[256];
    int t = threadIdx.x;
    float s = 0.f;
    for (int i = blockIdx.x * 256 + t; i < NROWS; i += gridDim.x * 256) s += npm[i];
    red[t] = s;
    __syncthreads();
    for (int off = 128; off > 0; off >>= 1) {
        if (t < off) red[t] += red[t + off];
        __syncthreads();
    }
    if (t == 0) partial[blockIdx.x] = red[0];
}
__global__ void k_reduce2(const float* __restrict__ partial, float* __restrict__ out, int nparts) {
    float s = ((int)threadIdx.x < nparts) ? partial[threadIdx.x] : 0.f;
#pragma unroll
    for (int off = 16; off; off >>= 1) s += __shfl_xor(s, off, 32);
    if (threadIdx.x == 0) out[0] = s * (1.0f / NROWS);
}

// ---------------------------------------------------------------------------
extern "C" void kernel_launch(void* const* d_in, const int* in_sizes, int n_in,
                              void* d_out, int out_size, void* d_ws, size_t ws_size,
                              hipStream_t stream) {
    (void)in_sizes; (void)n_in; (void)out_size; (void)ws_size;
    const float* x = (const float*)d_in[0];   // [8192,1024]
    const float* y = (const float*)d_in[1];   // [8192,1024]
    const float* W = (const float*)d_in[2];   // [1024,1024]
    const float* b = (const float*)d_in[3];   // [1024]
    float* out = (float*)d_out;

    char* ws = (char*)d_ws;
    size_t off = 0;
    auto take = [&](size_t bytes) -> char* {
        char* p = ws + off;
        off = (off + bytes + 255) & ~(size_t)255;
        return p;
    };
    float*  xpred  = (float*) take((size_t)NROWS * DIM * 4);   // 32 MB
    __bf16* ybf    = (__bf16*)take((size_t)NROWS * DIM * 2);   // 16 MB
    __bf16* wbf    = (__bf16*)take((size_t)DIM * DIM * 2);     //  2 MB
    __bf16* xnbf   = (__bf16*)take((size_t)NROWS * DIM * 2);   // 16 MB
    __bf16* xpnbf  = (__bf16*)take((size_t)NROWS * DIM * 2);   // 16 MB
    float*  pos    = (float*) take((size_t)NROWS * 4);
    float*  psum   = (float*) take((size_t)MSPLIT * NROWS * 4);
    float*  npm    = (float*) take((size_t)NROWS * 4);
    float*  partial= (float*) take(32 * 4);

    k_cvt_bf16<<<2048, 256, 0, stream>>>(y, ybf, NROWS * DIM);
    k_cvt_bf16<<<1024, 256, 0, stream>>>(W, wbf, DIM * DIM);

    // x_pred = y @ W^T   (bias folded into normalize)
    k_gemm1_wmma<<<(NROWS / 16) * (DIM / 16) / 8, 256, 0, stream>>>(ybf, wbf, xpred);

    k_rownorm_bf16<<<NROWS, 256, 0, stream>>>(x, b, 0, xnbf);
    k_rownorm_bf16<<<NROWS, 256, 0, stream>>>(xpred, b, 1, xpnbf);

    k_pos<<<NROWS / 8, 256, 0, stream>>>(xnbf, xpnbf, pos);

    k_gemm2_lse<<<dim3(NROWS / 128, MSPLIT), 256, 0, stream>>>(xnbf, xpnbf, psum);

    k_merge<<<NROWS / 256, 256, 0, stream>>>(psum, pos, npm);
    k_reduce1<<<32, 256, 0, stream>>>(npm, partial);
    k_reduce2<<<1, 32, 0, stream>>>(partial, out, 32);
}
